// EdgeEmbed_38044820308157
// MI455X (gfx1250) — matrix operand order repaired and checked by reading, not compiled, not run on hardware
//
#include <hip/hip_runtime.h>
#include <hip/hip_bf16.h>

#define NUM_EDGES 800000
#define NODE_DIM  128
#define EDGE_DIM  128
#define N_RADIAL  32

typedef __attribute__((ext_vector_type(16))) _Float16 v16h;
typedef __attribute__((ext_vector_type(8)))  _Float16 v8h;
typedef __attribute__((ext_vector_type(8)))  float    v8f;

// Scheduler pipeline hints (IGroupLP): 0x8 = MFMA/WMMA group, 0x100 = DS-read.
#if __has_builtin(__builtin_amdgcn_sched_group_barrier)
#define SGB(mask, size, id) __builtin_amdgcn_sched_group_barrier((mask), (size), (id))
#else
#define SGB(mask, size, id)
#endif

// ---------------------------------------------------------------------------
// Prep: W1T[n][k] = w_edge[k][n], W2T[n][k] = w_edge[128+k][n]  (f16, N-major)
//       WfT[n][c] = sum_d w_rbf[c][d] * w_edge[256+d][n]        (fused rbf path)
// ---------------------------------------------------------------------------
__global__ __launch_bounds__(128) void prep_weights_kernel(
    const float* __restrict__ w_rbf, const float* __restrict__ w_edge,
    _Float16* __restrict__ W1T, _Float16* __restrict__ W2T, _Float16* __restrict__ WfT)
{
    const int n = blockIdx.x;    // 0..127 output column
    const int k = threadIdx.x;   // 0..127
    W1T[n * 128 + k] = (_Float16)w_edge[k * 128 + n];
    W2T[n * 128 + k] = (_Float16)w_edge[(128 + k) * 128 + n];
    if (k < N_RADIAL) {
        float s = 0.0f;
        #pragma unroll 8
        for (int d = 0; d < 128; ++d)
            s += w_rbf[k * 128 + d] * w_edge[(256 + d) * 128 + n];
        WfT[n * N_RADIAL + k] = (_Float16)s;
    }
}

// ---------------------------------------------------------------------------
// Helpers
// ---------------------------------------------------------------------------
__device__ __forceinline__ v16h cvt_a(float4 f0, float4 f1, float4 f2, float4 f3) {
    v16h t;
    t[0]  = (_Float16)f0.x; t[1]  = (_Float16)f0.y; t[2]  = (_Float16)f0.z; t[3]  = (_Float16)f0.w;
    t[4]  = (_Float16)f1.x; t[5]  = (_Float16)f1.y; t[6]  = (_Float16)f1.z; t[7]  = (_Float16)f1.w;
    t[8]  = (_Float16)f2.x; t[9]  = (_Float16)f2.y; t[10] = (_Float16)f2.z; t[11] = (_Float16)f2.w;
    t[12] = (_Float16)f3.x; t[13] = (_Float16)f3.y; t[14] = (_Float16)f3.z; t[15] = (_Float16)f3.w;
    return t;
}

// A fragment (contiguous k order used by this code, B matches): lane {hi,m} holds row m,
// halves 0..15 = K[k0 + hi*16 + 0..15]
__device__ __forceinline__ v16h load_a_f32(const float* __restrict__ row, int k0, int hi) {
    const float* p = row + k0 + hi * 16;
    float4 f0 = *(const float4*)(p);
    float4 f1 = *(const float4*)(p + 4);
    float4 f2 = *(const float4*)(p + 8);
    float4 f3 = *(const float4*)(p + 12);
    return cvt_a(f0, f1, f2, f3);
}

// B fragment from 16 contiguous halves (two b128 loads; works for global & LDS)
__device__ __forceinline__ v16h load_b(const _Float16* p) {
    v8h lo = *(const v8h*)(p);
    v8h hh = *(const v8h*)(p + 8);
    return __builtin_shufflevector(lo, hh, 0,1,2,3,4,5,6,7,8,9,10,11,12,13,14,15);
}

// One K-step across all 8 n-tiles: batch-load 8 B fragments, then 8
// independent WMMAs (distinct accumulators -> no D->A/B hazards).
// When SCHED, request the pipeline: 2 B-pairs in flight, then WMMA/DS
// alternation -> each wmma waits for dscnt<=1 instead of a full drain.
template <bool SCHED>
__device__ __forceinline__ void kstep(v8f acc[8], v16h a,
                                      const _Float16* base, const int strideHalves) {
    v16h b[8];
    #pragma unroll
    for (int nt = 0; nt < 8; ++nt) b[nt] = load_b(base + nt * strideHalves);
    #pragma unroll
    for (int nt = 0; nt < 8; ++nt)
        acc[nt] = __builtin_amdgcn_wmma_f32_16x16x32_f16(false, a, false, b[nt],
                                                         (short)0, acc[nt], false, false);
    if (SCHED) {
        SGB(0x100, 4, 0);                   // prologue: two B pairs issued ahead
        #pragma unroll
        for (int i = 0; i < 7; ++i) {
            SGB(0x008, 1, 0);               // 1 wmma
            SGB(0x100, 2, 0);               // next B pair
        }
        SGB(0x008, 1, 0);                   // last wmma of this step
    }
}

// ---------------------------------------------------------------------------
// Main: out[e][n] = silu( hJ@W1 + hI@W2 + rbf@Wf + b )
// 8 waves/block, each wave = 16 edges x full 128 cols (8 accumulator tiles),
// 9 K-steps of wmma 16x16x32, ks-outer / nt-inner.
// W1/W2 staged in LDS in fragment-packed order:
//   region[mtx=0|1] : frag f = nt*4+kb, per-lane 16 halves at (f*32+lane)*16
// ---------------------------------------------------------------------------
__global__ __launch_bounds__(256) void edge_embed_kernel(
    const int*   __restrict__ x,
    const float* __restrict__ rbf,
    const int*   __restrict__ idx_i,
    const int*   __restrict__ idx_j,
    const float* __restrict__ embed_w,
    const float* __restrict__ b_edge,
    const _Float16* __restrict__ W1T,
    const _Float16* __restrict__ W2T,
    const _Float16* __restrict__ WfT,
    float* __restrict__ out)
{
    // 64 KB: [2 matrices][32 fragments][32 lanes][16 halves]
    __shared__ __align__(16) _Float16 smem[2 * 32 * 32 * 16];

    const int lane = threadIdx.x & 31;
    const int wave = threadIdx.x >> 5;
    const int m    = lane & 15;
    const int hi   = lane >> 4;

    // ---- Cooperative fill: global N-major -> LDS fragment-packed ----------
    {
        #pragma unroll
        for (int mtx = 0; mtx < 2; ++mtx) {
            const _Float16* wsrc = (mtx == 0) ? W1T : W2T;
            #pragma unroll
            for (int j = 0; j < 4; ++j) {
                int idx = j * 256 + threadIdx.x;           // 0..1023
                int f   = idx >> 5;                        // fragment 0..31
                int l   = idx & 31;                        // lane slot
                int n   = ((f >> 2) << 4) + (l & 15);      // output column
                int cb  = ((f & 3) << 5) + ((l >> 4) << 4);// K base for this slot
                const uint4* src = (const uint4*)(wsrc + n * 128 + cb);
                uint4 v0 = src[0];
                uint4 v1 = src[1];
                uint4* dst = (uint4*)(&smem[mtx * 16384 + idx * 16]);
                dst[0] = v0;
                dst[1] = v1;
            }
        }
    }
    __syncthreads();

    const int e_base = blockIdx.x * 128 + wave * 16;   // grid covers E exactly
    const int e      = e_base + m;

    const int ej = idx_j[e];
    const int ei = idx_i[e];
    const float* rowj = embed_w + (long)x[ej] * NODE_DIM;
    const float* rowi = embed_w + (long)x[ei] * NODE_DIM;
    const float* rrow = rbf + (long)e * N_RADIAL;

    // 8 accumulator tiles (full 128 output columns), bias preloaded once.
    v8f acc[8];
    #pragma unroll
    for (int nt = 0; nt < 8; ++nt) {
        const float bias = b_edge[nt * 16 + m];
        #pragma unroll
        for (int r = 0; r < 8; ++r) acc[nt][r] = bias;
    }

    // ---- 9 K-steps, A prefetched one step ahead ---------------------------
    v16h a = load_a_f32(rowj, 0, hi);

    #pragma unroll
    for (int ks = 0; ks < 4; ++ks) {            // K 0..127 : hJ @ W1
        v16h an = (ks < 3) ? load_a_f32(rowj, (ks + 1) * 32, hi)
                           : load_a_f32(rowi, 0, hi);
        kstep<true>(acc, a, &smem[ks * 512 + lane * 16], 2048);
        a = an;
    }
    #pragma unroll
    for (int ks = 0; ks < 4; ++ks) {            // K 128..255 : hI @ W2
        v16h an = (ks < 3) ? load_a_f32(rowi, (ks + 1) * 32, hi)
                           : load_a_f32(rrow, 0, hi);
        kstep<true>(acc, a, &smem[16384 + ks * 512 + lane * 16], 2048);
        a = an;
    }
    // K 256..287 : rbf @ Wf (B from global; 8 KB table, L0-resident)
    kstep<false>(acc, a, WfT + m * N_RADIAL + hi * 16, 512);

    // ---- SiLU + store. C/D layout: VGPR r -> row (r + 8*hi), col (nt*16+m)
    #pragma unroll
    for (int nt = 0; nt < 8; ++nt) {
        #pragma unroll
        for (int r = 0; r < 8; ++r) {
            float v = acc[nt][r];
            float s = v / (1.0f + __expf(-v));
            out[(long)(e_base + r + 8 * hi) * EDGE_DIM + nt * 16 + m] = s;
        }
    }
}

// ---------------------------------------------------------------------------
extern "C" void kernel_launch(void* const* d_in, const int* in_sizes, int n_in,
                              void* d_out, int out_size, void* d_ws, size_t ws_size,
                              hipStream_t stream) {
    const int*   x      = (const int*)  d_in[0];
    const float* rbf    = (const float*)d_in[1];
    const int*   idx_i  = (const int*)  d_in[2];
    const int*   idx_j  = (const int*)  d_in[3];
    const float* embed  = (const float*)d_in[4];
    const float* w_rbf  = (const float*)d_in[5];
    const float* w_edge = (const float*)d_in[6];
    const float* b_edge = (const float*)d_in[7];
    float* out = (float*)d_out;

    char* ws = (char*)d_ws;
    _Float16* W1T = (_Float16*)(ws);            // 128*128*2 = 32768 B
    _Float16* W2T = (_Float16*)(ws + 32768);    // 32768 B
    _Float16* WfT = (_Float16*)(ws + 65536);    // 128*32*2 = 8192 B

    prep_weights_kernel<<<128, 128, 0, stream>>>(w_rbf, w_edge, W1T, W2T, WfT);

    edge_embed_kernel<<<NUM_EDGES / 128, 256, 0, stream>>>(
        x, rbf, idx_i, idx_j, embed, b_edge, W1T, W2T, WfT, out);
}
